// MambaLikeAttention_3418793967826
// MI455X (gfx1250) — compile-verified, matrix operand order checked by reading, not run on hardware
//
#include <hip/hip_runtime.h>
#include <math.h>

#define EPS 1e-5f

typedef float v2f  __attribute__((ext_vector_type(2)));
typedef float v8f  __attribute__((ext_vector_type(8)));
typedef _Float16 v16h __attribute__((ext_vector_type(16)));

// Order-preserving float<->int mapping (involution), so integer atomicMax == float max.
__device__ __forceinline__ int f2ord(float x) {
    int i = __float_as_int(x);
    return i >= 0 ? i : (i ^ 0x7fffffff);
}
__device__ __forceinline__ float ord2f(int i) {
    int j = i >= 0 ? i : (i ^ 0x7fffffff);
    return __int_as_float(j);
}

// ---------------- BN param fold: scale = g/sqrt(v+eps), shift = b - m*scale ----
__global__ void k_bnparam(const float* __restrict__ g, const float* __restrict__ b,
                          const float* __restrict__ m, const float* __restrict__ v,
                          float* __restrict__ scale, float* __restrict__ shift) {
    int c = threadIdx.x;   // 256 threads, C=256
    float s = g[c] / sqrtf(v[c] + EPS);
    scale[c] = s;
    shift[c] = b[c] - m[c] * s;
}

// ---------------- init hw-max accumulator (b128 stores) ----------------------
__global__ void k_init(int* __restrict__ hwmax_i) {
    int i = blockIdx.x * blockDim.x + threadIdx.x;     // 0..32767 int4 groups
    int4 v; v.x = v.y = v.z = v.w = (int)0x80000000;   // <= f2ord of anything
    reinterpret_cast<int4*>(hwmax_i)[i] = v;
}

// ---------------- fused BN + triple pooling pass over x ----------------------
// Block = (b, 16-channel chunk). Each thread owns a 4x4 (h x w) micro-tile,
// streamed with float4 (global_load_b128) loads.
//  - hw pool (over C): register accumulators + int atomicMax / float4 partial sums
//  - cw pool (over H) & ch pool (over W): register partials in both directions,
//    combined via small padded LDS arrays (stride 65 / 17 -> conflict-free).
__global__ __launch_bounds__(256) void k_pool(
    const float* __restrict__ x,
    const float* __restrict__ scale, const float* __restrict__ shift,
    int* __restrict__ hwmax_i, float* __restrict__ hwsum_part,
    float* __restrict__ cwmax, float* __restrict__ cwmean,
    float* __restrict__ chmax, float* __restrict__ chmean) {
    __shared__ float cwpm[16 * 65];
    __shared__ float cwps[16 * 65];
    __shared__ float chpm[64 * 17];
    __shared__ float chps[64 * 17];

    const int tid = threadIdx.x;
    const int w4 = tid & 15;   // w = w4*4 + e
    const int q  = tid >> 4;   // 0..15, rows h = q*4 + i
    const int b = blockIdx.y;
    const int cb = blockIdx.x; // 0..15
    const int c0 = cb * 16;

    float hmax[4][4], hsum[4][4];
#pragma unroll
    for (int i = 0; i < 4; i++)
#pragma unroll
        for (int e = 0; e < 4; e++) { hmax[i][e] = -INFINITY; hsum[i][e] = 0.f; }

    for (int c = 0; c < 16; c++) {
        const int cc = c0 + c;
        const float s = scale[cc], sh = shift[cc];
        const float4* xp4 = reinterpret_cast<const float4*>(x + (b * 256 + cc) * 4096);
        float colmax[4], colsum[4], rowmax[4], rowsum[4];
#pragma unroll
        for (int e = 0; e < 4; e++) { colmax[e] = -INFINITY; colsum[e] = 0.f; }
#pragma unroll
        for (int i = 0; i < 4; i++) { rowmax[i] = -INFINITY; rowsum[i] = 0.f; }
#pragma unroll
        for (int i = 0; i < 4; i++) {
            int h = q * 4 + i;
            float4 xv = xp4[h * 16 + w4];
            float v[4] = { fmaf(xv.x, s, sh), fmaf(xv.y, s, sh),
                           fmaf(xv.z, s, sh), fmaf(xv.w, s, sh) };
#pragma unroll
            for (int e = 0; e < 4; e++) {
                hmax[i][e] = fmaxf(hmax[i][e], v[e]);
                hsum[i][e] += v[e];
                colmax[e] = fmaxf(colmax[e], v[e]);
                colsum[e] += v[e];
                rowmax[i] = fmaxf(rowmax[i], v[e]);
                rowsum[i] += v[e];
            }
        }
#pragma unroll
        for (int e = 0; e < 4; e++) {
            cwpm[q * 65 + w4 * 4 + e] = colmax[e];
            cwps[q * 65 + w4 * 4 + e] = colsum[e];
        }
#pragma unroll
        for (int i = 0; i < 4; i++) {
            chpm[(q * 4 + i) * 17 + w4] = rowmax[i];
            chps[(q * 4 + i) * 17 + w4] = rowsum[i];
        }
        __syncthreads();
        if (tid < 64) {  // cw: combine 16 q-partials per w
            float mx = -INFINITY, sm = 0.f;
#pragma unroll
            for (int g = 0; g < 16; g++) {
                mx = fmaxf(mx, cwpm[g * 65 + tid]);
                sm += cwps[g * 65 + tid];
            }
            int o = (b * 256 + cc) * 64 + tid;
            cwmax[o] = mx;
            cwmean[o] = sm * (1.f / 64.f);
        } else if (tid < 128) {  // ch: combine 16 w4-partials per h
            int h = tid - 64;
            float mx = -INFINITY, sm = 0.f;
#pragma unroll
            for (int g = 0; g < 16; g++) {
                mx = fmaxf(mx, chpm[h * 17 + g]);
                sm += chps[h * 17 + g];
            }
            int o = (b * 64 + h) * 256 + cc;
            chmax[o] = mx;
            chmean[o] = sm * (1.f / 64.f);
        }
        __syncthreads();
    }
#pragma unroll
    for (int i = 0; i < 4; i++) {
        int h = q * 4 + i;
#pragma unroll
        for (int e = 0; e < 4; e++) {
            int idx = (b * 64 + h) * 64 + w4 * 4 + e;
            atomicMax(&hwmax_i[idx], f2ord(hmax[i][e]));   // order-invariant
        }
        float4 sv = make_float4(hsum[i][0], hsum[i][1], hsum[i][2], hsum[i][3]);
        reinterpret_cast<float4*>(hwsum_part)[cb * 32768 + (b * 64 + h) * 16 + w4] = sv;
    }
}

// ---------------- finalize hw pool (b128) ------------------------------------
__global__ void k_hwfin(const int* __restrict__ hwmax_i, const float* __restrict__ part,
                        float* __restrict__ hwmax_f, float* __restrict__ hwmean_f) {
    int i = blockIdx.x * blockDim.x + threadIdx.x;  // 0..32767 float4 groups
    float4 s = make_float4(0.f, 0.f, 0.f, 0.f);
#pragma unroll
    for (int cb = 0; cb < 16; cb++) {
        float4 p = reinterpret_cast<const float4*>(part)[cb * 32768 + i];
        s.x += p.x; s.y += p.y; s.z += p.z; s.w += p.w;
    }
    int4 mi = reinterpret_cast<const int4*>(hwmax_i)[i];
    reinterpret_cast<float4*>(hwmax_f)[i] =
        make_float4(ord2f(mi.x), ord2f(mi.y), ord2f(mi.z), ord2f(mi.w));
    s.x *= (1.f / 256.f); s.y *= (1.f / 256.f);
    s.z *= (1.f / 256.f); s.w *= (1.f / 256.f);
    reinterpret_cast<float4*>(hwmean_f)[i] = s;
}

// ---------------- 7x7 conv(2->1) + BN + sigmoid via WMMA ---------------------
// Grid (Hd, B), 128 threads (4 full waves). Stage 7 halo rows of both channels
// in LDS, then each wave computes 16-output tiles: D = A(16 pix x 4 taps) x
// B(4 taps x 16, all columns = weights), accumulated over 25 chunks (98 taps,
// zero-padded). All 32 lanes active at every WMMA (EXEC all-ones).
__global__ __launch_bounds__(128) void k_conv_gate(
    const float* __restrict__ cmax, const float* __restrict__ cmean,
    const float* __restrict__ wconv,
    const float* __restrict__ g1p, const float* __restrict__ b1p,
    const float* __restrict__ m1p, const float* __restrict__ v1p,
    float* __restrict__ gate, int Hd, int Wd) {
    extern __shared__ float smem[];
    int* tapoff = (int*)smem;          // [128] LDS offset per tap (ci,dy,dx folded)
    float* wt = smem + 128;            // [128] weights, zero padded past 98
    float* sdata = smem + 256;         // [2][7][Wd+6] staged halo rows

    const int tid = threadIdx.x;
    const int hd = blockIdx.x;
    const int b = blockIdx.y;
    const int rs = Wd + 6;
    const int cs = 7 * rs;

    if (tid < 128) {
        if (tid < 98) {
            int ci = tid / 49, r = tid % 49, dy = r / 7, dx = r % 7;
            tapoff[tid] = ci * cs + dy * rs + dx;
            wt[tid] = wconv[tid];
        } else {
            tapoff[tid] = 0;
            wt[tid] = 0.f;
        }
    }
    const int bbase = b * Hd * Wd;
    for (int idx = tid; idx < 2 * cs; idx += 128) {
        int ci = idx / cs;
        int rr = idx - ci * cs;
        int dy = rr / rs;
        int j = rr - dy * rs;
        int r = hd + dy - 3, col = j - 3;
        float v = 0.f;
        if (r >= 0 && r < Hd && col >= 0 && col < Wd)
            v = (ci ? cmean : cmax)[bbase + r * Wd + col];
        sdata[idx] = v;
    }
    __syncthreads();

    const float s1 = g1p[0] / sqrtf(v1p[0] + EPS);
    const float t1 = b1p[0] - m1p[0] * s1;

    const int lane = tid & 31, wv = tid >> 5;
    const int m = lane & 15, half = lane >> 4;
    const int nt = Wd >> 4;
    const int gbase = (b * Hd + hd) * Wd;

    for (int t = wv; t < nt; t += 4) {
        const int w0 = t * 16;
        v8f acc = {0.f, 0.f, 0.f, 0.f, 0.f, 0.f, 0.f, 0.f};
#if defined(__HIP_DEVICE_COMPILE__)
#if __has_builtin(__builtin_amdgcn_wmma_f32_16x16x4_f32)
        // f32 16x16x4: A v0 = K(0|2), v1 = K(1|3) split by lane half; B mirrors.
        for (int j = 0; j < 25; j++) {
            int t0 = 4 * j + 2 * half;
            v2f A, B;
            A.x = sdata[tapoff[t0] + w0 + m];
            A.y = sdata[tapoff[t0 + 1] + w0 + m];
            B.x = wt[t0];
            B.y = wt[t0 + 1];
            acc = __builtin_amdgcn_wmma_f32_16x16x4_f32(false, A, false, B,
                                                        (short)0, acc, false, false);
        }
#else
        // Fallback: confirmed f16 builtin; 4 chunks of 32 taps (padded to 128).
        for (int j = 0; j < 4; j++) {
            int base = 32 * j;
            v16h A, B;
#pragma unroll
            for (int e = 0; e < 16; e++) {
                int kA = e + (e < 8 ? 0 : 8) + half * 8;
                A[e] = (_Float16)sdata[tapoff[base + kA] + w0 + m];
                B[e] = (_Float16)wt[base + e + half * 16];
            }
            acc = __builtin_amdgcn_wmma_f32_16x16x32_f16(false, A, false, B,
                                                         (short)0, acc, false, false);
        }
#endif
#endif
        // D[m][0]: lane 0 holds M=0..7 in acc[0..7], lane 16 holds M=8..15.
        if (lane == 0 || lane == 16) {
            int mb = half * 8;
#pragma unroll
            for (int r = 0; r < 8; r++) {
                float yy = fmaf(acc[r], s1, t1);
                gate[gbase + w0 + mb + r] = 1.f / (1.f + expf(-yy));
            }
        }
    }
}

// ---------------- final fused elementwise (float4) ---------------------------
__global__ __launch_bounds__(256) void k_final(
    const float* __restrict__ x,
    const float* __restrict__ scale, const float* __restrict__ shift,
    const float* __restrict__ ghw, const float* __restrict__ gcw,
    const float* __restrict__ gch, float* __restrict__ out) {
    int idx = blockIdx.x * 256 + threadIdx.x;  // over 8,388,608 float4's
    int w4 = idx & 15;
    int r = idx >> 4;
    int h = r & 63;
    r >>= 6;
    int c = r & 255;
    int b = r >> 8;

    float4 xv = reinterpret_cast<const float4*>(x)[idx];
    float s = scale[c], sh = shift[c];
    float4 a = reinterpret_cast<const float4*>(ghw)[(b * 64 + h) * 16 + w4];
    float4 d = reinterpret_cast<const float4*>(gcw)[(b * 256 + c) * 16 + w4];
    float e = gch[(b * 64 + h) * 256 + c];

    float4 o;
    o.x = xv.x + (a.x + d.x + e) * (1.f / 3.f) * fmaxf(fmaf(xv.x, s, sh), 0.f);
    o.y = xv.y + (a.y + d.y + e) * (1.f / 3.f) * fmaxf(fmaf(xv.y, s, sh), 0.f);
    o.z = xv.z + (a.z + d.z + e) * (1.f / 3.f) * fmaxf(fmaf(xv.z, s, sh), 0.f);
    o.w = xv.w + (a.w + d.w + e) * (1.f / 3.f) * fmaxf(fmaf(xv.w, s, sh), 0.f);
    reinterpret_cast<float4*>(out)[idx] = o;
}

extern "C" void kernel_launch(void* const* d_in, const int* in_sizes, int n_in,
                              void* d_out, int out_size, void* d_ws, size_t ws_size,
                              hipStream_t stream) {
    (void)in_sizes; (void)n_in; (void)out_size; (void)ws_size;
    const float* x = (const float*)d_in[0];
    const float* bng = (const float*)d_in[1];
    const float* bnb = (const float*)d_in[2];
    const float* bnm = (const float*)d_in[3];
    const float* bnv = (const float*)d_in[4];

    float* ws = (float*)d_ws;
    float* scale      = ws;                 // 256
    float* shift      = ws + 256;           // 256
    int*   hwmax_i    = (int*)(ws + 512);   // 131072 ints
    float* hwsum_part = ws + 131584;        // 16*131072
    float* hwmax_f    = ws + 2228736;       // 131072
    float* hwmean_f   = ws + 2359808;       // 131072
    float* cwmax      = ws + 2490880;       // 524288
    float* cwmean     = ws + 3015168;       // 524288
    float* chmax      = ws + 3539456;       // 524288
    float* chmean     = ws + 4063744;       // 524288
    float* ghw        = ws + 4588032;       // 131072
    float* gcw        = ws + 4719104;       // 524288
    float* gch        = ws + 5243392;       // 524288  (total ~22 MiB)

    k_bnparam<<<1, 256, 0, stream>>>(bng, bnb, bnm, bnv, scale, shift);
    k_init<<<128, 256, 0, stream>>>(hwmax_i);
    k_pool<<<dim3(16, 32), 256, 0, stream>>>(x, scale, shift, hwmax_i, hwsum_part,
                                             cwmax, cwmean, chmax, chmean);
    k_hwfin<<<128, 256, 0, stream>>>(hwmax_i, hwsum_part, hwmax_f, hwmean_f);

    const int smem64 = (256 + 14 * 70) * 4;
    const int smem256 = (256 + 14 * 262) * 4;
    // hw: image 64x64, gate[b][h][w]
    k_conv_gate<<<dim3(64, 32), 128, smem64, stream>>>(
        hwmax_f, hwmean_f, (const float*)d_in[5], (const float*)d_in[6],
        (const float*)d_in[7], (const float*)d_in[8], (const float*)d_in[9],
        ghw, 64, 64);
    // cw: image 256(c) x 64(w), gate[b][c][w]
    k_conv_gate<<<dim3(256, 32), 128, smem64, stream>>>(
        cwmax, cwmean, (const float*)d_in[10], (const float*)d_in[11],
        (const float*)d_in[12], (const float*)d_in[13], (const float*)d_in[14],
        gcw, 256, 64);
    // ch: image 64(h) x 256(c), gate[b][h][c]
    k_conv_gate<<<dim3(64, 32), 128, smem256, stream>>>(
        chmax, chmean, (const float*)d_in[15], (const float*)d_in[16],
        (const float*)d_in[17], (const float*)d_in[18], (const float*)d_in[19],
        gch, 64, 256);

    k_final<<<32768, 256, 0, stream>>>(x, scale, shift, ghw, gcw, gch, (float*)d_out);
}
